// DeepSeekMoE_69011534512397
// MI455X (gfx1250) — matrix operand order repaired and checked by reading, not compiled, not run on hardware
//
#include <hip/hip_runtime.h>

typedef unsigned short u16;
typedef __attribute__((ext_vector_type(16))) __bf16 v16bf;
typedef __attribute__((ext_vector_type(8)))  float  v8f;

#define T_TOK   2048
#define D_HID   1024
#define N_EXP   16
#define MID_R_  256
#define ROWPAD  8
#define LXS     (D_HID + ROWPAD)   /* 1032 elems; 2064B stride: 16B aligned, 16 mod 256 banks */

__device__ __forceinline__ u16 f2bf(float f) {
  unsigned x = __float_as_uint(f);
  unsigned r = (x + 0x7FFFu + ((x >> 16) & 1u)) >> 16;   // round-to-nearest-even
  return (u16)r;
}

union Frag { uint4 q[2]; v16bf v; };

// A-matrix 16x32 bf16 fragment (ISA 7.12.2): lane row = lane&15;
// lanes 0-15 hold K [k..k+7] & [k+16..k+23]; lanes 16-31 hold K [k+8..k+15] & [k+24..k+31]
__device__ __forceinline__ v16bf loadA(const u16* base, int k, int hi) {
  Frag f;
  f.q[0] = *(const uint4*)(base + k + hi * 8);
  f.q[1] = *(const uint4*)(base + k + 16 + hi * 8);
  return f.v;
}

// B-matrix 32x16 bf16 fragment from column-major storage (B^T row-major [N][K]):
// lane column = lane&15; lanes 0-15 hold K [k..k+7] & [k+16..k+23]; lanes 16-31 hold K [k+8..k+15] & [k+24..k+31]
__device__ __forceinline__ v16bf loadB(const u16* base, int k, int hi) {
  Frag f;
  f.q[0] = *(const uint4*)(base + k + hi * 8);
  f.q[1] = *(const uint4*)(base + k + 16 + hi * 8);
  return f.v;
}

// ---------------- fp32 -> bf16 convert ----------------
__global__ __launch_bounds__(256) void cvt_bf16(const float* __restrict__ s,
                                                u16* __restrict__ d, int n) {
  int i = blockIdx.x * 256 + threadIdx.x;
  if (i < n) d[i] = f2bf(s[i]);
}

// ---------------- fp32 [E][K][N] -> bf16 [E][N][K] transpose-convert ----------------
__global__ __launch_bounds__(256) void transpose_bf16(const float* __restrict__ src,
                                                      u16* __restrict__ dst,
                                                      int K, int N) {
  __shared__ float tile[32][33];
  int e = blockIdx.z;
  const float* s = src + (size_t)e * K * N;
  u16* d = dst + (size_t)e * N * K;
  int n0 = blockIdx.x * 32, k0 = blockIdx.y * 32;
  int tx = threadIdx.x & 31, ty = threadIdx.x >> 5;
  for (int i = ty; i < 32; i += 8) {
    int k = k0 + i, n = n0 + tx;
    tile[i][tx] = (k < K && n < N) ? s[(size_t)k * N + n] : 0.f;
  }
  __syncthreads();
  for (int i = ty; i < 32; i += 8) {
    int n = n0 + i, k = k0 + tx;
    if (n < N && k < K) d[(size_t)n * K + k] = f2bf(tile[tx][i]);
  }
}

// ---------------- router: logits -> top2 -> renormalized weights ----------------
__global__ __launch_bounds__(256) void router_kernel(const float* __restrict__ x,
                                                     const float* __restrict__ rw,
                                                     int* __restrict__ tki,
                                                     float* __restrict__ tkw) {
  __shared__ float lg[16][16];
  int tid = threadIdx.x;
  int half = tid >> 4;          // token slot within block (16 tokens/block)
  int e = tid & 15;
  int t = blockIdx.x * 16 + half;
  const float* xr = x + (size_t)t * D_HID;
  float acc = 0.f;
  for (int k = 0; k < D_HID; k++) acc += xr[k] * rw[k * N_EXP + e];
  lg[half][e] = acc;
  __syncthreads();
  if (tid < 16) {
    int tt = blockIdx.x * 16 + tid;
    float l1 = -3.4e38f, l2 = -3.4e38f;
    int i1 = 0, i2 = 0;
    for (int j = 0; j < N_EXP; j++) {
      float v = lg[tid][j];
      if (v > l1) { l2 = l1; i2 = i1; l1 = v; i1 = j; }
      else if (v > l2) { l2 = v; i2 = j; }
    }
    float w1 = 1.f / (1.f + __expf(l2 - l1));   // == p1/(p1+p2)
    tki[tt * 2 + 0] = i1; tki[tt * 2 + 1] = i2;
    tkw[tt * 2 + 0] = w1; tkw[tt * 2 + 1] = 1.f - w1;
  }
}

// ---------------- deterministic per-expert gather lists ----------------
__global__ void build_lists(const int* __restrict__ tki, const float* __restrict__ tkw,
                            int* __restrict__ pairs, float* __restrict__ pw,
                            int* __restrict__ counts) {
  int e = threadIdx.x;
  if (e >= N_EXP) return;
  int c = 0;
  for (int t = 0; t < T_TOK; t++) {
    for (int s = 0; s < 2; s++) {
      if (tki[t * 2 + s] == e) {
        pairs[(size_t)e * T_TOK + c] = t * 2 + s;
        pw[(size_t)e * T_TOK + c] = tkw[t * 2 + s];
        c++;
      }
    }
  }
  counts[e] = c;
}

// ---------------- fused SwiGLU expert FFN: out_row = (silu(xWg)*xWu*scale) Wd ------
// Weights are bf16, pre-transposed: wgT/wuT = [E][M][D], wdT = [E][D][M].
// pairs==nullptr -> identity rows (shared expert path), scale 1, out row = token.
__global__ __launch_bounds__(256) void expert_ffn(
    const u16* __restrict__ xb,
    const u16* __restrict__ wgT, const u16* __restrict__ wuT, const u16* __restrict__ wdT,
    const int* __restrict__ pairs, const float* __restrict__ pw,
    const int* __restrict__ counts,
    int ntok, int M, float* __restrict__ outp) {
  __shared__ u16 lx[32][LXS];
  __shared__ u16 lh[32][LXS];
  __shared__ float lscale[32];
  __shared__ int lrow[32];
  __shared__ int ltok[32];

  const int e = blockIdx.y;
  const u16* wg = wgT + (size_t)e * M * D_HID;
  const u16* wu = wuT + (size_t)e * M * D_HID;
  const u16* wd = wdT + (size_t)e * D_HID * M;
  const int cnt = counts ? counts[e] : ntok;
  const int row0 = blockIdx.x * 32;
  if (row0 >= cnt) return;                       // block-uniform

  const int tid = threadIdx.x;
  if (tid < 32) {
    int r = row0 + tid;
    int p = -1, tok = 0; float sc = 0.f;
    if (r < cnt) {
      if (pairs) { p = pairs[(size_t)e * T_TOK + r]; sc = pw[(size_t)e * T_TOK + r]; tok = p >> 1; }
      else       { p = r; tok = r; sc = 1.f; }
    }
    lrow[tid] = p; ltok[tid] = tok; lscale[tid] = sc;
  }
  __syncthreads();

  // stage 32 gathered token rows of bf16 x into LDS
  for (int i = tid; i < 32 * (D_HID / 8); i += 256) {
    int row = i >> 7, seg = i & 127;
    uint4 v = *(const uint4*)(xb + (size_t)ltok[row] * D_HID + seg * 8);
    *(uint4*)(&lx[row][seg * 8]) = v;
  }
  __syncthreads();

  const int wv = tid >> 5, lane = tid & 31;
  const int ln = lane & 15, hi = lane >> 4;
  const int NC = M >> 4;

  // phase 1: g/u projections share the A fragment; h = silu(g)*u*scale -> lh (bf16)
  for (int t = wv; t < 2 * NC; t += 8) {
    int trow = t / NC, tcol = t % NC;
    const u16* arow = &lx[trow * 16 + ln][0];
    const u16* bg = wg + (size_t)(tcol * 16 + ln) * D_HID;
    const u16* bu = wu + (size_t)(tcol * 16 + ln) * D_HID;
    v8f cg = {}; v8f cu = {};
    for (int k = 0; k < D_HID; k += 32) {
      v16bf a = loadA(arow, k, hi);
      v16bf g = loadB(bg, k, hi);
      v16bf u = loadB(bu, k, hi);
      cg = __builtin_amdgcn_wmma_f32_16x16x32_bf16(false, a, false, g, (short)0, cg, false, false);
      cu = __builtin_amdgcn_wmma_f32_16x16x32_bf16(false, a, false, u, (short)0, cu, false, false);
    }
#pragma unroll
    for (int r = 0; r < 8; r++) {
      int rr = trow * 16 + (hi ? 8 + r : r);     // C/D layout: lanes16-31 hold M=8..15
      float g = cg[r], u = cu[r];
      float h = (g / (1.f + __expf(-g))) * u * lscale[rr];
      lh[rr][tcol * 16 + ln] = f2bf(h);
    }
  }
  __syncthreads();

  // phase 2: out = h * Wd   (K = M, N = 1024)
  for (int t = wv; t < 2 * 64; t += 8) {
    int trow = t >> 6, tcol = t & 63;
    const u16* arow = &lh[trow * 16 + ln][0];
    const u16* brow = wd + (size_t)(tcol * 16 + ln) * M;
    v8f c = {};
    for (int k = 0; k < M; k += 32) {
      v16bf a = loadA(arow, k, hi);
      v16bf b = loadB(brow, k, hi);
      c = __builtin_amdgcn_wmma_f32_16x16x32_bf16(false, a, false, b, (short)0, c, false, false);
    }
#pragma unroll
    for (int r = 0; r < 8; r++) {
      int rr = trow * 16 + (hi ? 8 + r : r);
      int p = lrow[rr];
      if (p >= 0) outp[(size_t)p * D_HID + tcol * 16 + ln] = c[r];
    }
  }
}

// ---------------- combine: out += routed slot0 + slot1 ----------------
__global__ __launch_bounds__(256) void combine_kernel(float* __restrict__ out,
                                                      const float* __restrict__ rout) {
  int i = blockIdx.x * 256 + threadIdx.x;
  if (i >= T_TOK * D_HID) return;
  int t = i >> 10, d = i & 1023;
  out[i] += rout[(size_t)(2 * t) * D_HID + d] + rout[(size_t)(2 * t + 1) * D_HID + d];
}

extern "C" void kernel_launch(void* const* d_in, const int* in_sizes, int n_in,
                              void* d_out, int out_size, void* d_ws, size_t ws_size,
                              hipStream_t stream) {
  (void)in_sizes; (void)n_in; (void)out_size; (void)ws_size;
  const float* x    = (const float*)d_in[0];
  const float* rw   = (const float*)d_in[1];
  const float* wg_r = (const float*)d_in[2];
  const float* wu_r = (const float*)d_in[3];
  const float* wd_r = (const float*)d_in[4];
  const float* wg_s = (const float*)d_in[5];
  const float* wu_s = (const float*)d_in[6];
  const float* wd_s = (const float*)d_in[7];
  float* out = (float*)d_out;

  char* ws = (char*)d_ws;
  size_t off = 0;
  auto alloc = [&](size_t bytes) -> char* {
    char* p = ws + off; off += (bytes + 255) & ~(size_t)255; return p;
  };
  u16*   xb   = (u16*)alloc((size_t)T_TOK * D_HID * 2);                 // 4 MB
  u16*   wgsT = (u16*)alloc((size_t)D_HID * D_HID * 2);                 // 2 MB  [2][512][1024] == [1024][1024]
  u16*   wusT = (u16*)alloc((size_t)D_HID * D_HID * 2);                 // 2 MB
  u16*   wdsT = (u16*)alloc((size_t)D_HID * D_HID * 2);                 // 2 MB  [1024(d)][1024(m)]
  u16*   wgrT = (u16*)alloc((size_t)N_EXP * MID_R_ * D_HID * 2);        // 8 MB  [16][256][1024]
  u16*   wurT = (u16*)alloc((size_t)N_EXP * MID_R_ * D_HID * 2);        // 8 MB
  u16*   wdrT = (u16*)alloc((size_t)N_EXP * D_HID * MID_R_ * 2);        // 8 MB  [16][1024][256]
  int*   tki  = (int*)alloc((size_t)T_TOK * 2 * 4);
  float* tkw  = (float*)alloc((size_t)T_TOK * 2 * 4);
  int*   pairs= (int*)alloc((size_t)N_EXP * T_TOK * 4);
  float* pw   = (float*)alloc((size_t)N_EXP * T_TOK * 4);
  int*   cnts = (int*)alloc((size_t)N_EXP * 4);
  float* rout = (float*)alloc((size_t)T_TOK * 2 * D_HID * 4);           // 16 MB

  // --- precision conversion / weight re-layout (L2-resident afterwards) ---
  cvt_bf16<<<(T_TOK * D_HID + 255) / 256, 256, 0, stream>>>(x, xb, T_TOK * D_HID);
  transpose_bf16<<<dim3(512 / 32, 1024 / 32, 2),  256, 0, stream>>>(wg_s, wgsT, 1024, 512);
  transpose_bf16<<<dim3(512 / 32, 1024 / 32, 2),  256, 0, stream>>>(wu_s, wusT, 1024, 512);
  transpose_bf16<<<dim3(1024 / 32, 1024 / 32, 1), 256, 0, stream>>>(wd_s, wdsT, 1024, 1024);
  transpose_bf16<<<dim3(256 / 32, 1024 / 32, 16), 256, 0, stream>>>(wg_r, wgrT, 1024, 256);
  transpose_bf16<<<dim3(256 / 32, 1024 / 32, 16), 256, 0, stream>>>(wu_r, wurT, 1024, 256);
  transpose_bf16<<<dim3(1024 / 32, 256 / 32, 16), 256, 0, stream>>>(wd_r, wdrT, 256, 1024);

  // --- routing (fp32 for numerics) + deterministic gather ---
  router_kernel<<<T_TOK / 16, 256, 0, stream>>>(x, rw, tki, tkw);
  build_lists<<<1, 32, 0, stream>>>(tki, tkw, pairs, pw, cnts);

  // --- shared experts: one fused M=1024 FFN, plain-stores d_out ---
  expert_ffn<<<dim3(T_TOK / 32, 1), 256, 0, stream>>>(
      xb, wgsT, wusT, wdsT, nullptr, nullptr, nullptr, T_TOK, 1024, out);

  // --- routed experts: gathered top-2 rows, each (token,slot) row stored once ---
  expert_ffn<<<dim3(T_TOK / 32, N_EXP), 256, 0, stream>>>(
      xb, wgrT, wurT, wdrT, pairs, pw, cnts, T_TOK, MID_R_, rout);

  // --- out += slot0 + slot1 ---
  combine_kernel<<<(T_TOK * D_HID + 255) / 256, 256, 0, stream>>>(out, rout);
}
